// VQ_25443386262157
// MI455X (gfx1250) — matrix operand order repaired and checked by reading, not compiled, not run on hardware
//
#include <hip/hip_runtime.h>
#include <cstdint>
#include <cstddef>

#define N_  16384
#define E_  512
#define C_  128
#define KC_ 2048
#define M_  4
#define H1_ 128
#define H2_ 256

typedef __attribute__((ext_vector_type(16))) __bf16 v16bf;
typedef __attribute__((ext_vector_type(8)))  float  v8f;

// ---------------------------------------------------------------------------
// CDNA5 async global->LDS staging (ASYNCcnt path). Generic LDS pointer's low
// 32 bits are the LDS byte address (aperture layout, ISA 10.2).
// NBYTES must be a multiple of 4096 (256 threads x 16B).
// ---------------------------------------------------------------------------
template <int NBYTES>
__device__ __forceinline__ void stage_async(void* lds_dst, const void* gsrc, int tid) {
  uint32_t lbase = (uint32_t)(uintptr_t)lds_dst + (uint32_t)(tid << 4);
  uint64_t gbase = (uint64_t)(uintptr_t)gsrc + (uint64_t)(tid << 4);
  constexpr int PER = NBYTES / 4096;
#pragma unroll
  for (int r = 0; r < PER; ++r) {
    asm volatile("global_load_async_to_lds_b128 %0, %1, off"
                 :: "v"(lbase + (uint32_t)(r * 4096)),
                    "v"(gbase + (uint64_t)(r * 4096)) : "memory");
  }
  asm volatile("s_wait_asynccnt 0" ::: "memory");  // caller must __syncthreads()
}

// ---------------------------------------------------------------------------
// Tensor Data Mover: 1-D contiguous tile load (n8 = size in 8-byte units).
// D# layout per CDNA5 ISA ch.8 (group0: count|lds|global|type, group1: dims).
// ---------------------------------------------------------------------------
#if __has_builtin(__builtin_amdgcn_tensor_load_to_lds)
#define HAVE_TDM 1
typedef __attribute__((ext_vector_type(4))) unsigned int u32x4;
typedef __attribute__((ext_vector_type(8))) int i32x8;
typedef __attribute__((ext_vector_type(4))) int i32x4;

__device__ __forceinline__ void tdm_load_1d(void* lds_dst, const void* gsrc,
                                            unsigned n8) {
  unsigned lds = (unsigned)(uintptr_t)lds_dst;
  unsigned long long ga = (unsigned long long)(uintptr_t)gsrc;
  u32x4 g0;
  g0[0] = 1u;                                   // count=1 (valid user D#)
  g0[1] = lds;                                  // lds_addr
  g0[2] = (unsigned)ga;                         // global_addr[31:0]
  g0[3] = (unsigned)(ga >> 32) | (2u << 30);    // global_addr[56:32] | type=2
  i32x8 g1;
  g1[0] = (int)(3u << 16);                      // data_size = 8 bytes
  g1[1] = (int)(n8 << 16);                      // tensor_dim0[15:0]
  g1[2] = (int)((1u << 16) | (n8 >> 16));       // tensor_dim1=1 | dim0[31:16]
  g1[3] = (int)(n8 << 16);                      // tile_dim0 (n8 < 65536)
  g1[4] = 1;                                    // tile_dim1 = 1
  g1[5] = (int)n8;                              // tensor_dim0_stride[31:0]
  g1[6] = 0;
  g1[7] = 0;
  i32x4 z4 = {0, 0, 0, 0};
#if __clang_major__ >= 23
  i32x8 z8 = {0, 0, 0, 0, 0, 0, 0, 0};
  __builtin_amdgcn_tensor_load_to_lds(g0, g1, z4, z4, z8, 0);
#else
  __builtin_amdgcn_tensor_load_to_lds(g0, g1, z4, z4, 0);
#endif
}
__device__ __forceinline__ void tdm_wait() {
  __builtin_amdgcn_s_wait_tensorcnt(0);
}
#else
#define HAVE_TDM 0
#endif

// Stage a contiguous tile: TDM from wave 0 if available, else async copies.
template <int NBYTES>
__device__ __forceinline__ void stage_tile(void* lds_dst, const void* gsrc, int tid) {
#if HAVE_TDM
  if (tid < 32) {               // uniform per wave; TDM ignores EXEC
    if (tid == 0) tdm_load_1d(lds_dst, gsrc, NBYTES / 8);
    tdm_wait();
  }
#else
  stage_async<NBYTES>(lds_dst, gsrc, tid);
#endif
}

// ---------------------------------------------------------------------------
// WMMA fragment builders (wave32); A and B both hold k = 16*half + j in element j.
// A 16x32 bf16 (MxK) from row-major [16][lda]: 2x ds_load_b128 per lane.
// B 32x16 bf16 fragment-major [tile][lane][16]: contiguous 32B per lane ->
//   2x coalesced global_load_b128 per wave (1KB/tile, streams from L2).
// C/D f32: VGPR i -> M = i + (lane>=16 ? 8 : 0), N = lane&15.
// ---------------------------------------------------------------------------
__device__ __forceinline__ v16bf load_A(const __bf16* A, int lda, int k0, int lane) {
  int half = (lane >> 4) & 1, row = lane & 15;
  const __bf16* r = A + (size_t)row * lda + k0 + half * 16;
  v16bf a;
#pragma unroll
  for (int j = 0; j < 16; ++j)
    a[j] = r[j];
  return a;
}

__device__ __forceinline__ v16bf load_B_sw(const __bf16* base, int tile, int lane) {
  return *(const v16bf*)(base + (((size_t)tile << 5) + lane) * 16);
}

__device__ __forceinline__ v8f wmma_bf16(v16bf a, v16bf b, v8f c) {
  return __builtin_amdgcn_wmma_f32_16x16x32_bf16(false, a, false, b,
                                                 (short)0, c, false, false);
}

// ---------------------------------------------------------------------------
// Prep kernels
// ---------------------------------------------------------------------------
__global__ void cvt_bf16_kernel(const float* __restrict__ src,
                                __bf16* __restrict__ dst, size_t n) {
  size_t i = blockIdx.x * (size_t)blockDim.x + threadIdx.x;
  size_t stride = (size_t)gridDim.x * blockDim.x;
  for (; i < n; i += stride) dst[i] = (__bf16)src[i];
}

// Swizzle [batch][Krows][Ncols] f32 -> fragment-major bf16
// out[b][kc][nt][lane][j] = W[b][kc*32 + half*16 + j][nt*16 + col]
__global__ void swizzle_w_kernel(const float* __restrict__ W, __bf16* __restrict__ out,
                                 int Krows, int Ncols, int batch) {
  size_t total = (size_t)batch * Krows * Ncols;
  size_t i = blockIdx.x * (size_t)blockDim.x + threadIdx.x;
  size_t stride = (size_t)gridDim.x * blockDim.x;
  int ntiles = Ncols >> 4, kchunks = Krows >> 5;
  for (; i < total; i += stride) {
    int j = (int)(i & 15);
    int lane = (int)((i >> 4) & 31);
    size_t rest = i >> 9;
    int nt = (int)(rest % ntiles);
    size_t rest2 = rest / ntiles;
    int kc = (int)(rest2 % kchunks);
    int b = (int)(rest2 / kchunks);
    int half = lane >> 4, col = lane & 15;
    int k = kc * 32 + half * 16 + j;
    int n = nt * 16 + col;
    out[i] = (__bf16)W[((size_t)b * Krows + k) * Ncols + n];
  }
}

// Codebook as B-matrix (rows=c (C_), cols=k (KC_)), fragment-major:
// out[m][cc][kt][lane][j] = cb[m][kt*16+col][cc*32+half*16+j]
__global__ void swizzle_cb_kernel(const float* __restrict__ cb,
                                  __bf16* __restrict__ out) {
  size_t total = (size_t)M_ * C_ * KC_;
  size_t i = blockIdx.x * (size_t)blockDim.x + threadIdx.x;
  size_t stride = (size_t)gridDim.x * blockDim.x;
  for (; i < total; i += stride) {
    int j = (int)(i & 15);
    int lane = (int)((i >> 4) & 31);
    size_t rest = i >> 9;
    int kt = (int)(rest % (KC_ / 16));
    size_t rest2 = rest / (KC_ / 16);
    int cc = (int)(rest2 % (C_ / 32));
    int m = (int)(rest2 / (C_ / 32));
    int half = lane >> 4, col = lane & 15;
    int c = cc * 32 + half * 16 + j;
    int k = kt * 16 + col;
    out[i] = (__bf16)cb[((size_t)m * KC_ + k) * C_ + c];
  }
}

// one block per (m,k) row: cb_sq[m][k] = sum_c cb^2
__global__ __launch_bounds__(128) void cbsq_kernel(const float* __restrict__ cb,
                                                   float* __restrict__ cbsq) {
  int mk = blockIdx.x;
  int c = threadIdx.x;
  float v = cb[(size_t)mk * C_ + c];
  __shared__ float red[128];
  red[c] = v * v;
  __syncthreads();
#pragma unroll
  for (int s = 64; s > 0; s >>= 1) {
    if (c < s) red[c] += red[c + s];
    __syncthreads();
  }
  if (c == 0) cbsq[mk] = red[0];
}

// fold (linear-bias + eval BatchNorm) into per-channel scale/offset
__global__ void fold_bn_kernel(const float* __restrict__ bias, const float* __restrict__ g,
                               const float* __restrict__ beta, const float* __restrict__ mean,
                               const float* __restrict__ var, float* __restrict__ s,
                               float* __restrict__ t, int n) {
  int i = blockIdx.x * blockDim.x + threadIdx.x;
  if (i < n) {
    float sc = g[i] * rsqrtf(var[i] + 1e-5f);
    s[i] = sc;
    t[i] = (bias[i] - mean[i]) * sc + beta[i];
  }
}

// ---------------------------------------------------------------------------
// Encoder: per (m, 64-row tile), 256 threads = 8 waves. 4 row-subtiles per
// wave reuse each B fragment 4x from registers; no barriers in k-loops.
// ---------------------------------------------------------------------------
__global__ __launch_bounds__(256) void encoder_kernel(
    const __bf16* __restrict__ xbf,
    const __bf16* __restrict__ W1, const __bf16* __restrict__ W2,
    const __bf16* __restrict__ W3,                        // swizzled
    const float* __restrict__ s1, const float* __restrict__ t1,
    const float* __restrict__ s2, const float* __restrict__ t2,
    const float* __restrict__ b3,
    float* __restrict__ ze_out, __bf16* __restrict__ ze_bf)
{
  __shared__ __align__(16) __bf16 xt[64 * E_];    // 64KB
  __shared__ __align__(16) __bf16 h1s[64 * H1_];  // 16KB
  __shared__ __align__(16) __bf16 h2s[64 * H2_];  // 32KB

  int m = blockIdx.x / (N_ / 64);
  int tile = blockIdx.x % (N_ / 64);
  int row0 = tile * 64;
  int tid = threadIdx.x, wave = tid >> 5, lane = tid & 31;
  int half = (lane >> 4) & 1, col = lane & 15;

  stage_tile<64 * E_ * 2>(xt, xbf + (size_t)row0 * E_, tid);
  __syncthreads();

  // ---- GEMM1: [64,512]x[512,128], wave w -> col tile w ----
  {
    v8f acc[4] = {};
    const __bf16* Wm = W1 + (size_t)m * E_ * H1_;
    for (int kc = 0; kc < E_ / 32; ++kc) {
      v16bf b = load_B_sw(Wm + (size_t)kc * 32 * H1_, wave, lane);
#pragma unroll
      for (int r = 0; r < 4; ++r) {
        v16bf a = load_A(xt + (size_t)r * 16 * E_, E_, kc * 32, lane);
        acc[r] = wmma_bf16(a, b, acc[r]);
      }
    }
    int ch = wave * 16 + col;
    float s = s1[m * H1_ + ch], t = t1[m * H1_ + ch];
#pragma unroll
    for (int r = 0; r < 4; ++r)
#pragma unroll
      for (int i = 0; i < 8; ++i) {
        float v = acc[r][i] * s + t;
        h1s[(r * 16 + i + half * 8) * H1_ + ch] = (__bf16)(v > 0.f ? v : 0.f);
      }
  }
  __syncthreads();

  // ---- GEMM2: [64,128]x[128,256], wave w -> col tiles w, w+8 ----
  {
    v8f acc[2][4] = {};
    const __bf16* Wm = W2 + (size_t)m * H1_ * H2_;
    for (int kc = 0; kc < H1_ / 32; ++kc) {
      const __bf16* ck = Wm + (size_t)kc * 32 * H2_;
      v16bf b0 = load_B_sw(ck, wave, lane);
      v16bf b1 = load_B_sw(ck, wave + 8, lane);
#pragma unroll
      for (int r = 0; r < 4; ++r) {
        v16bf a = load_A(h1s + (size_t)r * 16 * H1_, H1_, kc * 32, lane);
        acc[0][r] = wmma_bf16(a, b0, acc[0][r]);
        acc[1][r] = wmma_bf16(a, b1, acc[1][r]);
      }
    }
#pragma unroll
    for (int tt = 0; tt < 2; ++tt) {
      int ch = wave * 16 + tt * 128 + col;
      float s = s2[m * H2_ + ch], t = t2[m * H2_ + ch];
#pragma unroll
      for (int r = 0; r < 4; ++r)
#pragma unroll
        for (int i = 0; i < 8; ++i) {
          float v = acc[tt][r][i] * s + t;
          h2s[(r * 16 + i + half * 8) * H2_ + ch] = (__bf16)(v > 0.f ? v : 0.f);
        }
    }
  }
  __syncthreads();

  // ---- GEMM3: [64,256]x[256,128] -> ze ----
  {
    v8f acc[4] = {};
    const __bf16* Wm = W3 + (size_t)m * H2_ * C_;
    for (int kc = 0; kc < H2_ / 32; ++kc) {
      v16bf b = load_B_sw(Wm + (size_t)kc * 32 * C_, wave, lane);
#pragma unroll
      for (int r = 0; r < 4; ++r) {
        v16bf a = load_A(h2s + (size_t)r * 16 * H2_, H2_, kc * 32, lane);
        acc[r] = wmma_bf16(a, b, acc[r]);
      }
    }
    int ch = wave * 16 + col;
    float bias = b3[m * C_ + ch];
#pragma unroll
    for (int r = 0; r < 4; ++r)
#pragma unroll
      for (int i = 0; i < 8; ++i) {
        float v = acc[r][i] + bias;
        size_t o = ((size_t)m * N_ + row0 + r * 16 + i + half * 8) * C_ + ch;
        ze_out[o] = v;
        ze_bf[o] = (__bf16)v;
      }
  }
}

// ---------------------------------------------------------------------------
// VQ: argmin_k( cb_sq[k] - 2*<ze, cb_k> ), 32-row tiles, 2 row-subtiles.
// ---------------------------------------------------------------------------
__device__ __forceinline__ uint32_t ordkey(float f) {
  uint32_t u = __float_as_uint(f);
  return (u & 0x80000000u) ? ~u : (u | 0x80000000u);
}

__global__ __launch_bounds__(256) void vq_kernel(
    const __bf16* __restrict__ ze_bf,   // [M,N,C]
    const __bf16* __restrict__ cbsw,    // [M][C/32][K/16][32][16]
    const float* __restrict__ cbsq,     // [M,K]
    const float* __restrict__ cb,       // [M,K,C] fp32
    float* __restrict__ ce_out)         // [M,N,C]
{
  __shared__ __align__(16) __bf16 zet[32 * C_];
  __shared__ unsigned long long best[32];

  int m = blockIdx.x / (N_ / 32);
  int tile = blockIdx.x % (N_ / 32);
  int row0 = tile * 32;
  int tid = threadIdx.x, wave = tid >> 5, lane = tid & 31;
  int half = (lane >> 4) & 1, col = lane & 15;

  if (tid < 32) best[tid] = ~0ULL;
  stage_tile<32 * C_ * 2>(zet, ze_bf + ((size_t)m * N_ + row0) * C_, tid);
  __syncthreads();

  // hoist all A fragments (2 row-subtiles x 4 K-chunks)
  v16bf af[2][C_ / 32];
#pragma unroll
  for (int r = 0; r < 2; ++r)
#pragma unroll
    for (int cc = 0; cc < C_ / 32; ++cc)
      af[r][cc] = load_A(zet + (size_t)r * 16 * C_, C_, cc * 32, lane);

  const __bf16* cbm = cbsw + (size_t)m * C_ * KC_;
  float bv[2][8];
  int bi[2][8];
#pragma unroll
  for (int r = 0; r < 2; ++r)
#pragma unroll
    for (int i = 0; i < 8; ++i) { bv[r][i] = 3.4028235e38f; bi[r][i] = 0; }

  for (int kt = wave; kt < KC_ / 16; kt += 8) {   // 16 col tiles per wave
    v8f acc0 = {}, acc1 = {};
#pragma unroll
    for (int cc = 0; cc < C_ / 32; ++cc) {
      v16bf b = load_B_sw(cbm + (size_t)cc * (KC_ / 16) * 512, kt, lane);
      acc0 = wmma_bf16(af[0][cc], b, acc0);
      acc1 = wmma_bf16(af[1][cc], b, acc1);
    }
    float cq = cbsq[m * KC_ + kt * 16 + col];
#pragma unroll
    for (int i = 0; i < 8; ++i) {
      float d0 = cq - 2.f * acc0[i];
      if (d0 < bv[0][i]) { bv[0][i] = d0; bi[0][i] = kt * 16 + col; }
      float d1 = cq - 2.f * acc1[i];
      if (d1 < bv[1][i]) { bv[1][i] = d1; bi[1][i] = kt * 16 + col; }
    }
  }
#pragma unroll
  for (int r = 0; r < 2; ++r)
#pragma unroll
    for (int i = 0; i < 8; ++i) {
      unsigned long long pk =
          ((unsigned long long)ordkey(bv[r][i]) << 32) | (unsigned)bi[r][i];
      atomicMin(&best[r * 16 + i + half * 8], pk);  // ds_min_u64
    }
  __syncthreads();

  for (int idx = tid; idx < 32 * C_; idx += 256) {
    int r = idx >> 7, c = idx & 127;
    int k = (int)(best[r] & 0xFFFFFFFFULL);
    ce_out[((size_t)m * N_ + row0 + r) * C_ + c] =
        cb[((size_t)m * KC_ + k) * C_ + c];
  }
}

// ---------------------------------------------------------------------------
// Decoder: zq = sum_m ce ; C->H2->H1->E with folded BN+relu; 64-row tiles.
// ---------------------------------------------------------------------------
__global__ __launch_bounds__(256) void decoder_kernel(
    const float* __restrict__ ce,
    const __bf16* __restrict__ dW1, const __bf16* __restrict__ dW2,
    const __bf16* __restrict__ dW3,                        // swizzled
    const float* __restrict__ ds1, const float* __restrict__ dt1,
    const float* __restrict__ ds2, const float* __restrict__ dt2,
    const float* __restrict__ db3, float* __restrict__ xhat)
{
  __shared__ __align__(16) __bf16 zq[64 * C_];    // 16KB
  __shared__ __align__(16) __bf16 d1s[64 * H2_];  // 32KB
  __shared__ __align__(16) __bf16 d2s[64 * H1_];  // 16KB

  int row0 = blockIdx.x * 64;
  int tid = threadIdx.x, wave = tid >> 5, lane = tid & 31;
  int half = (lane >> 4) & 1, col = lane & 15;

  for (int idx = tid; idx < 64 * C_; idx += 256) {
    int r = idx >> 7, c = idx & 127;
    float ssum = 0.f;
#pragma unroll
    for (int m = 0; m < M_; ++m)
      ssum += ce[((size_t)m * N_ + row0 + r) * C_ + c];
    zq[idx] = (__bf16)ssum;
  }
  __syncthreads();

  // ---- GEMM1: [64,128]x[128,256], col tiles w, w+8 ----
  {
    v8f acc[2][4] = {};
    for (int kc = 0; kc < C_ / 32; ++kc) {
      const __bf16* ck = dW1 + (size_t)kc * 32 * H2_;
      v16bf b0 = load_B_sw(ck, wave, lane);
      v16bf b1 = load_B_sw(ck, wave + 8, lane);
#pragma unroll
      for (int r = 0; r < 4; ++r) {
        v16bf a = load_A(zq + (size_t)r * 16 * C_, C_, kc * 32, lane);
        acc[0][r] = wmma_bf16(a, b0, acc[0][r]);
        acc[1][r] = wmma_bf16(a, b1, acc[1][r]);
      }
    }
#pragma unroll
    for (int tt = 0; tt < 2; ++tt) {
      int ch = wave * 16 + tt * 128 + col;
      float s = ds1[ch], t = dt1[ch];
#pragma unroll
      for (int r = 0; r < 4; ++r)
#pragma unroll
        for (int i = 0; i < 8; ++i) {
          float v = acc[tt][r][i] * s + t;
          d1s[(r * 16 + i + half * 8) * H2_ + ch] = (__bf16)(v > 0.f ? v : 0.f);
        }
    }
  }
  __syncthreads();

  // ---- GEMM2: [64,256]x[256,128] ----
  {
    v8f acc[4] = {};
    for (int kc = 0; kc < H2_ / 32; ++kc) {
      v16bf b = load_B_sw(dW2 + (size_t)kc * 32 * H1_, wave, lane);
#pragma unroll
      for (int r = 0; r < 4; ++r) {
        v16bf a = load_A(d1s + (size_t)r * 16 * H2_, H2_, kc * 32, lane);
        acc[r] = wmma_bf16(a, b, acc[r]);
      }
    }
    int ch = wave * 16 + col;
    float s = ds2[ch], t = dt2[ch];
#pragma unroll
    for (int r = 0; r < 4; ++r)
#pragma unroll
      for (int i = 0; i < 8; ++i) {
        float v = acc[r][i] * s + t;
        d2s[(r * 16 + i + half * 8) * H1_ + ch] = (__bf16)(v > 0.f ? v : 0.f);
      }
  }
  __syncthreads();

  // ---- GEMM3: [64,128]x[128,512] -> x_hat, col tiles w+8j, j=0..3 ----
#pragma unroll
  for (int jp = 0; jp < 2; ++jp) {
    v8f acc[2][4] = {};
    for (int kc = 0; kc < C_ / 32; ++kc) {
      const __bf16* ck = dW3 + (size_t)kc * 32 * E_;
      v16bf b0 = load_B_sw(ck, wave + 8 * (jp * 2 + 0), lane);
      v16bf b1 = load_B_sw(ck, wave + 8 * (jp * 2 + 1), lane);
#pragma unroll
      for (int r = 0; r < 4; ++r) {
        v16bf a = load_A(d2s + (size_t)r * 16 * C_, C_, kc * 32, lane);
        acc[0][r] = wmma_bf16(a, b0, acc[0][r]);
        acc[1][r] = wmma_bf16(a, b1, acc[1][r]);
      }
    }
#pragma unroll
    for (int tt = 0; tt < 2; ++tt) {
      int ch = (wave + 8 * (jp * 2 + tt)) * 16 + col;
      float bias = db3[ch];
#pragma unroll
      for (int r = 0; r < 4; ++r)
#pragma unroll
        for (int i = 0; i < 8; ++i)
          xhat[(size_t)(row0 + r * 16 + i + half * 8) * E_ + ch] =
              acc[tt][r][i] + bias;
    }
  }
}

// ---------------------------------------------------------------------------
// Host launcher
// ---------------------------------------------------------------------------
extern "C" void kernel_launch(void* const* d_in, const int* in_sizes, int n_in,
                              void* d_out, int out_size, void* d_ws, size_t ws_size,
                              hipStream_t stream) {
  const float* x      = (const float*)d_in[0];
  const float* eW1    = (const float*)d_in[1];
  const float* eb1    = (const float*)d_in[2];
  const float* bn1g   = (const float*)d_in[3];
  const float* bn1b   = (const float*)d_in[4];
  const float* bn1m   = (const float*)d_in[5];
  const float* bn1v   = (const float*)d_in[6];
  const float* eW2    = (const float*)d_in[7];
  const float* eb2    = (const float*)d_in[8];
  const float* bn2g   = (const float*)d_in[9];
  const float* bn2b   = (const float*)d_in[10];
  const float* bn2m   = (const float*)d_in[11];
  const float* bn2v   = (const float*)d_in[12];
  const float* eW3    = (const float*)d_in[13];
  const float* eb3    = (const float*)d_in[14];
  const float* cb     = (const float*)d_in[15];
  const float* dW1f   = (const float*)d_in[16];
  const float* db1    = (const float*)d_in[17];
  const float* dbn1g  = (const float*)d_in[18];
  const float* dbn1b  = (const float*)d_in[19];
  const float* dbn1m  = (const float*)d_in[20];
  const float* dbn1v  = (const float*)d_in[21];
  const float* dW2f   = (const float*)d_in[22];
  const float* db2    = (const float*)d_in[23];
  const float* dbn2g  = (const float*)d_in[24];
  const float* dbn2b  = (const float*)d_in[25];
  const float* dbn2m  = (const float*)d_in[26];
  const float* dbn2v  = (const float*)d_in[27];
  const float* dW3f   = (const float*)d_in[28];
  const float* db3    = (const float*)d_in[29];

  // d_out layout: x_hat [N,E] | ze [M,N,C] | ce [M,N,C]
  float* xhat   = (float*)d_out;
  float* ze_out = xhat + (size_t)N_ * E_;
  float* ce_out = ze_out + (size_t)M_ * N_ * C_;

  char* ws = (char*)d_ws;
  size_t off = 0;
  auto carve = [&](size_t bytes) -> void* {
    void* p = ws + off;
    off += (bytes + 255) & ~(size_t)255;
    return p;
  };
  __bf16* xbf   = (__bf16*)carve((size_t)N_ * E_ * 2);
  __bf16* W1sw  = (__bf16*)carve((size_t)M_ * E_ * H1_ * 2);
  __bf16* W2sw  = (__bf16*)carve((size_t)M_ * H1_ * H2_ * 2);
  __bf16* W3sw  = (__bf16*)carve((size_t)M_ * H2_ * C_ * 2);
  __bf16* cbsw  = (__bf16*)carve((size_t)M_ * C_ * KC_ * 2);
  __bf16* dW1s  = (__bf16*)carve((size_t)C_ * H2_ * 2);
  __bf16* dW2s  = (__bf16*)carve((size_t)H2_ * H1_ * 2);
  __bf16* dW3s  = (__bf16*)carve((size_t)H1_ * E_ * 2);
  __bf16* ze_bf = (__bf16*)carve((size_t)M_ * N_ * C_ * 2);
  float*  cbsq  = (float*)carve((size_t)M_ * KC_ * 4);
  float*  s1    = (float*)carve((size_t)M_ * H1_ * 4);
  float*  t1    = (float*)carve((size_t)M_ * H1_ * 4);
  float*  s2    = (float*)carve((size_t)M_ * H2_ * 4);
  float*  t2    = (float*)carve((size_t)M_ * H2_ * 4);
  float*  ds1   = (float*)carve((size_t)H2_ * 4);
  float*  dt1   = (float*)carve((size_t)H2_ * 4);
  float*  ds2   = (float*)carve((size_t)H1_ * 4);
  float*  dt2   = (float*)carve((size_t)H1_ * 4);

  auto blocks_for = [](size_t n) {
    size_t b = (n + 255) / 256;
    return (int)(b > 4096 ? 4096 : b);
  };

  cvt_bf16_kernel<<<blocks_for((size_t)N_ * E_), 256, 0, stream>>>(
      x, xbf, (size_t)N_ * E_);

  swizzle_w_kernel<<<blocks_for((size_t)M_ * E_ * H1_), 256, 0, stream>>>(
      eW1, W1sw, E_, H1_, M_);
  swizzle_w_kernel<<<blocks_for((size_t)M_ * H1_ * H2_), 256, 0, stream>>>(
      eW2, W2sw, H1_, H2_, M_);
  swizzle_w_kernel<<<blocks_for((size_t)M_ * H2_ * C_), 256, 0, stream>>>(
      eW3, W3sw, H2_, C_, M_);
  swizzle_w_kernel<<<blocks_for((size_t)C_ * H2_), 256, 0, stream>>>(
      dW1f, dW1s, C_, H2_, 1);
  swizzle_w_kernel<<<blocks_for((size_t)H2_ * H1_), 256, 0, stream>>>(
      dW2f, dW2s, H2_, H1_, 1);
  swizzle_w_kernel<<<blocks_for((size_t)H1_ * E_), 256, 0, stream>>>(
      dW3f, dW3s, H1_, E_, 1);
  swizzle_cb_kernel<<<blocks_for((size_t)M_ * C_ * KC_), 256, 0, stream>>>(cb, cbsw);

  cbsq_kernel<<<M_ * KC_, 128, 0, stream>>>(cb, cbsq);

  fold_bn_kernel<<<(M_ * H1_ + 255) / 256, 256, 0, stream>>>(eb1, bn1g, bn1b, bn1m, bn1v, s1, t1, M_ * H1_);
  fold_bn_kernel<<<(M_ * H2_ + 255) / 256, 256, 0, stream>>>(eb2, bn2g, bn2b, bn2m, bn2v, s2, t2, M_ * H2_);
  fold_bn_kernel<<<(H2_ + 255) / 256, 256, 0, stream>>>(db1, dbn1g, dbn1b, dbn1m, dbn1v, ds1, dt1, H2_);
  fold_bn_kernel<<<(H1_ + 255) / 256, 256, 0, stream>>>(db2, dbn2g, dbn2b, dbn2m, dbn2v, ds2, dt2, H1_);

  encoder_kernel<<<M_ * (N_ / 64), 256, 0, stream>>>(
      xbf, W1sw, W2sw, W3sw, s1, t1, s2, t2, eb3, ze_out, ze_bf);

  vq_kernel<<<M_ * (N_ / 32), 256, 0, stream>>>(ze_bf, cbsw, cbsq, cb, ce_out);

  decoder_kernel<<<N_ / 64, 256, 0, stream>>>(
      ce_out, dW1s, dW2s, dW3s, ds1, dt1, ds2, dt2, db3, xhat);
}
